// Compressor_74646531604795
// MI455X (gfx1250) — compile-verified, matrix operand order checked by reading, not run on hardware
//
#include <hip/hip_runtime.h>
#include <math.h>
#include <stdint.h>

// ---------------------------------------------------------------------------
// Perceiver resampler forward for MI455X (gfx1250, wave32, WMMA + TDM).
// Roofline: ~1.37 TFLOP vs ~768MB fp32 weights -> memory-bound (~33us floor
// @ 23.3 TB/s). Therefore:
//   * weights stream fp32 from HBM exactly once per block-tile, DMA'd by the
//     Tensor Data Mover into LDS (tensor_load_to_lds / s_wait_tensorcnt)
//   * one cooperative fp32->bf16 transpose per slab, fragments then read as
//     contiguous ds_load_b128
//   * all matmuls via v_wmma_f32_16x16x32_bf16 (fp32 accumulate)
//   * flash attention: 16-query tile per wave, online softmax, P through LDS
// ---------------------------------------------------------------------------

#define DIM    2048
#define HEADS  16
#define DHD    128
#define NQ     512
#define SEQ    2048
#define BATCH  4
#define FFD    8192
#define NLAYER 4

typedef __attribute__((ext_vector_type(16))) __bf16 v16bf;
typedef __attribute__((ext_vector_type(8)))  float  v8f;
typedef __attribute__((ext_vector_type(4)))  unsigned int u32x4;
typedef __attribute__((ext_vector_type(4)))  int i32x4;
typedef __attribute__((ext_vector_type(8)))  int i32x8;

union FragU { v16bf v; unsigned int u[8]; uint4 q[2]; };

#if defined(__has_builtin)
#  if __has_builtin(__builtin_amdgcn_tensor_load_to_lds) && \
      __has_builtin(__builtin_amdgcn_s_wait_tensorcnt)
#    define HAVE_TDM 1
#  endif
#endif
#ifndef HAVE_TDM
#  define HAVE_TDM 0
#endif

__device__ __forceinline__ unsigned int f2bf(float x) {
  unsigned int b = __float_as_uint(x);
  return (b + 0x7fffu + ((b >> 16) & 1u)) >> 16;   // round-to-nearest-even
}
// 16-bit A/B fragment K-offset for VGPR v, lane-half hi (ISA 7.12.2):
// v in 0..3 -> K = 2v + 8*hi ; v in 4..7 -> K = 16 + 2(v-4) + 8*hi
__device__ __forceinline__ int frag_kd(int v, int hi) {
  return ((v & 4) << 2) + hi * 8 + ((v & 3) << 1);
}
__device__ __forceinline__ v8f wmma_bf16(const FragU& a, const FragU& b, v8f c) {
  return __builtin_amdgcn_wmma_f32_16x16x32_bf16(
      false, a.v, false, b.v, (short)0, c, false, false);
}

#if HAVE_TDM
// Issue a TDM load of a 32x128 fp32 tile (row stride = n_stride elements)
// starting at gptr into LDS offset lds_off. D# per CDNA5 ISA §8.3/8.4.
__device__ __forceinline__ void tdm_load_b_tile(unsigned int lds_off,
                                                const void* gptr,
                                                unsigned int n_stride,
                                                unsigned int tensor_rows) {
  unsigned long long ga = (unsigned long long)(uintptr_t)gptr;
  u32x4 g0;
  g0[0] = 1u;                                         // count=1 (valid), user mode
  g0[1] = lds_off;                                    // lds_addr (bytes)
  g0[2] = (unsigned int)(ga & 0xffffffffull);         // global_addr[31:0]
  g0[3] = (unsigned int)((ga >> 32) & 0x01ffffffull)  // global_addr[56:32]
        | (2u << 30);                                 // type = 2 ("image")
  i32x8 g1;
  g1[0] = (int)(2u << 16);                            // data_size = 4B
  g1[1] = (int)((n_stride & 0xffffu) << 16);          // tensor_dim0[15:0]
  g1[2] = (int)((n_stride >> 16) & 0xffffu)
        | (int)((tensor_rows & 0xffffu) << 16);       // dim0 hi | tensor_dim1 lo
  g1[3] = (int)((tensor_rows >> 16) & 0xffffu)
        | (int)(128u << 16);                          // dim1 hi | tile_dim0 = 128
  g1[4] = 32;                                         // tile_dim1 = 32, tile_dim2 = 0
  g1[5] = (int)n_stride;                              // tensor_dim0_stride lo32
  g1[6] = 0;                                          // stride hi | dim1_stride lo
  g1[7] = 0;
  i32x4 z4 = (i32x4)0;
#if defined(__clang_major__) && (__clang_major__ >= 23)
  i32x8 z8 = (i32x8)0;
  __builtin_amdgcn_tensor_load_to_lds(g0, g1, z4, z4, z8, 0);
#else
  __builtin_amdgcn_tensor_load_to_lds(g0, g1, z4, z4, 0);
#endif
}
#endif

// ---------------------------------------------------------------------------
// LayerNorm over rows of width 2048 (one 256-thread block per row).
// Optionally fuses "+ pos_emb[row % SEQ]" and writes bf16 or fp32.
// ---------------------------------------------------------------------------
template<bool ADD_POS, bool OUT_F32>
__global__ __launch_bounds__(256)
void ln_kernel(const float* __restrict__ in, const float* __restrict__ pos,
               const float* __restrict__ g, const float* __restrict__ bta,
               void* __restrict__ outp)
{
  const int row = blockIdx.x;
  const float* x = in + (size_t)row * DIM;
  const float* p = ADD_POS ? (pos + (size_t)(row & (SEQ - 1)) * DIM) : nullptr;

  float v[8];
  float s = 0.f, s2 = 0.f;
  #pragma unroll
  for (int j = 0; j < 8; ++j) {
    int idx = threadIdx.x + j * 256;
    float t = x[idx];
    if (ADD_POS) t += p[idx];
    v[j] = t; s += t; s2 += t * t;
  }
  #pragma unroll
  for (int off = 16; off > 0; off >>= 1) {
    s  += __shfl_down(s,  off, 32);
    s2 += __shfl_down(s2, off, 32);
  }
  __shared__ float rs[8], rs2[8];
  const int wid = threadIdx.x >> 5, lane = threadIdx.x & 31;
  if (lane == 0) { rs[wid] = s; rs2[wid] = s2; }
  __syncthreads();
  if (threadIdx.x == 0) {
    float a = 0.f, b2 = 0.f;
    #pragma unroll
    for (int i = 0; i < 8; ++i) { a += rs[i]; b2 += rs2[i]; }
    rs[0] = a; rs2[0] = b2;
  }
  __syncthreads();
  const float mean = rs[0] * (1.f / DIM);
  const float var  = rs2[0] * (1.f / DIM) - mean * mean;
  const float rstd = rsqrtf(var + 1e-5f);
  #pragma unroll
  for (int j = 0; j < 8; ++j) {
    int idx = threadIdx.x + j * 256;
    float y = (v[j] - mean) * rstd * g[idx] + bta[idx];
    if (OUT_F32) ((float*)outp)[(size_t)row * DIM + idx] = y;
    else ((unsigned short*)outp)[(size_t)row * DIM + idx] = (unsigned short)f2bf(y);
  }
}

// ---------------------------------------------------------------------------
// GEMM: C[M,N] = scale * (A_bf16[M,K] @ B_f32[K,N]) with epilogue.
// Block 256 thr = 8 waves (2x4); wave tile 64x32 = 4x2 WMMA tiles; BK=32.
// Weight slab (32x128 fp32) DMA'd into LDS by TDM, converted once to a
// K-contiguous bf16 layout, fragments then read as two b128 per lane.
// ---------------------------------------------------------------------------
enum { EPI_BF16 = 0, EPI_SILU = 1, EPI_F32ADD = 2, EPI_KV = 3 };

template<int EPI>
__global__ __launch_bounds__(256)
void gemm_bf16w(const unsigned short* __restrict__ A, int lda,
                const float* __restrict__ Bw, int N, int K, float scale,
                unsigned short* __restrict__ outb,
                float* __restrict__ outf, const float* __restrict__ resid,
                unsigned short* __restrict__ kbuf, unsigned short* __restrict__ vtbuf)
{
  __shared__ float          ldsB[32 * 128];     // raw fp32 slab from TDM
  __shared__ unsigned short ldsBT[128 * 32];    // bf16, [col][k] K-contiguous

  const int lane = threadIdx.x & 31;
  const int wave = threadIdx.x >> 5;
  const int wm = wave & 1, wn = wave >> 1;
  const int r16 = lane & 15, hi = lane >> 4;
  const int rowBase  = blockIdx.y * 128 + wm * 64;
  const int colBase0 = blockIdx.x * 128;          // block col origin
  const int colBase  = colBase0 + wn * 32;        // wave col origin

  v8f acc[4][2] = {};
  for (int k0 = 0; k0 < K; k0 += 32) {
    __syncthreads();   // prior iteration's LDS reads complete
#if HAVE_TDM
    if (wave == 0) {
      tdm_load_b_tile((unsigned int)(uintptr_t)(void*)&ldsB[0],
                      Bw + (size_t)k0 * N + colBase0,
                      (unsigned int)N, (unsigned int)(K - k0));
      __builtin_amdgcn_s_wait_tensorcnt((short)0);
    }
#else
    {
      const float* src = Bw + (size_t)k0 * N + colBase0;
      #pragma unroll
      for (int j = 0; j < 16; ++j) {
        int e = threadIdx.x + j * 256;
        ldsB[e] = src[(size_t)(e >> 7) * N + (e & 127)];
      }
    }
#endif
    if (k0 + 32 < K)  // global_prefetch_b8 next weight slab toward L2
      __builtin_prefetch(Bw + (size_t)(k0 + 32) * N + colBase0 + threadIdx.x, 0, 1);
    __syncthreads();   // slab in LDS

    // fp32 [k][n] -> bf16 [n][k]; each weight converted once per block
    #pragma unroll
    for (int j = 0; j < 16; ++j) {
      int e = threadIdx.x + j * 256;
      ldsBT[(e & 127) * 32 + (e >> 7)] = (unsigned short)f2bf(ldsB[e]);
    }
    __syncthreads();   // transposed bf16 ready

    FragU a[4];
    #pragma unroll
    for (int i = 0; i < 4; ++i) {
      const unsigned short* ap =
          A + (size_t)(rowBase + i * 16 + r16) * lda + k0 + hi * 8;
      a[i].q[0] = *(const uint4*)ap;          // K = hi*8 .. hi*8+7
      a[i].q[1] = *(const uint4*)(ap + 16);   // K = 16+hi*8 .. 16+hi*8+7
    }
    FragU b[2];
    #pragma unroll
    for (int j = 0; j < 2; ++j) {
      const unsigned short* bp = &ldsBT[(wn * 32 + j * 16 + r16) * 32 + hi * 8];
      b[j].q[0] = *(const uint4*)bp;
      b[j].q[1] = *(const uint4*)(bp + 16);
    }
    #pragma unroll
    for (int i = 0; i < 4; ++i)
      #pragma unroll
      for (int j = 0; j < 2; ++j)
        acc[i][j] = wmma_bf16(a[i], b[j], acc[i][j]);
  }

  // C/D layout: VGPR r -> row (r + 8*hi), col = lane&15 within 16x16 tile.
  #pragma unroll
  for (int i = 0; i < 4; ++i) {
    #pragma unroll
    for (int j = 0; j < 2; ++j) {
      const int col = colBase + j * 16 + r16;
      #pragma unroll
      for (int r = 0; r < 8; ++r) {
        const int row = rowBase + i * 16 + hi * 8 + r;
        float val = acc[i][j][r] * scale;
        if constexpr (EPI == EPI_BF16) {
          outb[(size_t)row * N + col] = (unsigned short)f2bf(val);
        } else if constexpr (EPI == EPI_SILU) {
          float sv = val / (1.f + __expf(-val));
          outb[(size_t)row * N + col] = (unsigned short)f2bf(sv);
        } else if constexpr (EPI == EPI_F32ADD) {
          outf[(size_t)row * N + col] = val + resid[(size_t)row * N + col];
        } else { // EPI_KV: split columns into K [b,h,s,d] and V^T [b,h,d,s]
          const int bb = row >> 11;           // row = b*SEQ + s
          const int ss = row & (SEQ - 1);
          if (col < DIM) {
            const int hh = col >> 7, dd = col & (DHD - 1);
            kbuf[((size_t)(bb * HEADS + hh) * SEQ + ss) * DHD + dd] =
                (unsigned short)f2bf(val);
          } else {
            const int c2 = col - DIM;
            const int hh = c2 >> 7, dd = c2 & (DHD - 1);
            vtbuf[((size_t)(bb * HEADS + hh) * DHD + dd) * SEQ + ss] =
                (unsigned short)f2bf(val);
          }
        }
      }
    }
  }
}

// ---------------------------------------------------------------------------
// Flash cross-attention. One wave = one (b, h, 16-query tile). 4 waves/block.
// Keys processed 32 at a time: QK^T (8 WMMA) -> online softmax -> P via LDS
// (C-layout -> A-layout transpose) -> P @ V (8 WMMA). V is pre-transposed.
// ---------------------------------------------------------------------------
__global__ __launch_bounds__(128)
void attn_flash(const unsigned short* __restrict__ qb,
                const unsigned short* __restrict__ kb,
                const unsigned short* __restrict__ vtb,
                unsigned short* __restrict__ ob)
{
  __shared__ unsigned short pl[4][16][32];   // per-wave P tile (16x32 bf16)
  const int lane = threadIdx.x & 31;
  const int wave = threadIdx.x >> 5;
  const int r16 = lane & 15, hi = lane >> 4;
  const int tile = blockIdx.x * 4 + wave;
  const int qt = tile & 31;
  const int h  = (tile >> 5) & (HEADS - 1);
  const int b  = tile >> 9;

  const unsigned int* Q32 = (const unsigned int*)qb;
  const unsigned int* K32 =
      (const unsigned int*)kb + (size_t)(b * HEADS + h) * SEQ * (DHD / 2);
  const unsigned int* V32 =
      (const unsigned int*)vtb + (size_t)(b * HEADS + h) * DHD * (SEQ / 2);
  const int qrow0 = b * NQ + qt * 16;

  FragU qf[4];
  #pragma unroll
  for (int c = 0; c < 4; ++c) {
    size_t base = (size_t)(qrow0 + r16) * (DIM / 2) + h * (DHD / 2) + c * 16;
    #pragma unroll
    for (int v = 0; v < 8; ++v)
      qf[c].u[v] = Q32[base + (frag_kd(v, hi) >> 1)];
  }

  float m[8], lsum[8];
  #pragma unroll
  for (int r = 0; r < 8; ++r) { m[r] = -1e30f; lsum[r] = 0.f; }
  v8f o[8] = {};

  for (int kc = 0; kc < SEQ / 32; ++kc) {
    v8f s0 = {}, s1 = {};
    #pragma unroll
    for (int c = 0; c < 4; ++c) {
      FragU kf0, kf1;
      size_t b0 = (size_t)(kc * 32 + r16) * (DHD / 2) + c * 16;
      size_t b1 = (size_t)(kc * 32 + 16 + r16) * (DHD / 2) + c * 16;
      #pragma unroll
      for (int v = 0; v < 8; ++v) {
        const int kh = frag_kd(v, hi) >> 1;
        kf0.u[v] = K32[b0 + kh];
        kf1.u[v] = K32[b1 + kh];
      }
      s0 = wmma_bf16(qf[c], kf0, s0);
      s1 = wmma_bf16(qf[c], kf1, s1);
    }
    float alpha[8];
    #pragma unroll
    for (int r = 0; r < 8; ++r) {
      float rowm = fmaxf(s0[r], s1[r]);
      #pragma unroll
      for (int off = 1; off < 16; off <<= 1)
        rowm = fmaxf(rowm, __shfl_xor(rowm, off, 32));
      const float nm = fmaxf(m[r], rowm);
      alpha[r] = __expf(m[r] - nm);
      m[r] = nm;
      const float p0 = __expf(s0[r] - nm);
      const float p1 = __expf(s1[r] - nm);
      s0[r] = p0; s1[r] = p1;
      float rsum = p0 + p1;
      #pragma unroll
      for (int off = 1; off < 16; off <<= 1)
        rsum += __shfl_xor(rsum, off, 32);
      lsum[r] = lsum[r] * alpha[r] + rsum;
    }
    #pragma unroll
    for (int nt = 0; nt < 8; ++nt)
      #pragma unroll
      for (int r = 0; r < 8; ++r)
        o[nt][r] *= alpha[r];

    __syncthreads();                      // prior reads done (uniform count)
    #pragma unroll
    for (int r = 0; r < 8; ++r) {
      pl[wave][hi * 8 + r][r16]      = (unsigned short)f2bf(s0[r]);
      pl[wave][hi * 8 + r][16 + r16] = (unsigned short)f2bf(s1[r]);
    }
    __syncthreads();                      // P visible in A-layout order

    FragU pf;
    const unsigned short* pr = &pl[wave][r16][hi * 8];
    pf.q[0] = *(const uint4*)pr;
    pf.q[1] = *(const uint4*)(pr + 16);

    #pragma unroll
    for (int nt = 0; nt < 8; ++nt) {
      FragU vf;
      size_t vb = (size_t)(nt * 16 + r16) * (SEQ / 2) + kc * 16;
      #pragma unroll
      for (int v = 0; v < 8; ++v)
        vf.u[v] = V32[vb + (frag_kd(v, hi) >> 1)];
      o[nt] = wmma_bf16(pf, vf, o[nt]);
    }
  }

  #pragma unroll
  for (int nt = 0; nt < 8; ++nt)
    #pragma unroll
    for (int r = 0; r < 8; ++r) {
      const float val = o[nt][r] / lsum[r];
      const size_t idx =
          (size_t)(qrow0 + hi * 8 + r) * DIM + h * DHD + nt * 16 + r16;
      ob[idx] = (unsigned short)f2bf(val);
    }
}

// ---------------------------------------------------------------------------
// Host orchestration
// ---------------------------------------------------------------------------
extern "C" void kernel_launch(void* const* d_in, const int* in_sizes, int n_in,
                              void* d_out, int out_size, void* d_ws, size_t ws_size,
                              hipStream_t stream) {
  (void)in_sizes; (void)n_in; (void)out_size; (void)ws_size;
  const float* latents = (const float*)d_in[0];
  const float* seg     = (const float*)d_in[1];
  const float* pos     = (const float*)d_in[2];
  const float* nx_g    = (const float*)d_in[3];
  const float* nx_b    = (const float*)d_in[4];
  const float* nl_g    = (const float*)d_in[5];
  const float* nl_b    = (const float*)d_in[6];
  const float* Wq      = (const float*)d_in[7];
  const float* Wkv     = (const float*)d_in[8];
  const float* Wo      = (const float*)d_in[9];
  const float* fln_g   = (const float*)d_in[10];
  const float* fln_b   = (const float*)d_in[11];
  const float* W1      = (const float*)d_in[12];
  const float* W2      = (const float*)d_in[13];
  const float* fn_g    = (const float*)d_in[14];
  const float* fn_b    = (const float*)d_in[15];
  float* out = (float*)d_out;

  char* ws = (char*)d_ws;
  const size_t MB = 1u << 20;
  float*          lat   = (float*)(ws);                    // 16 MB fp32 residual
  unsigned short* xln   = (unsigned short*)(ws + 16 * MB); // 32 MB
  unsigned short* latln = (unsigned short*)(ws + 48 * MB); //  8 MB
  unsigned short* qb    = (unsigned short*)(ws + 56 * MB); //  8 MB
  unsigned short* kbuf  = (unsigned short*)(ws + 64 * MB); // 32 MB
  unsigned short* vtbuf = (unsigned short*)(ws + 96 * MB); // 32 MB
  unsigned short* ao    = (unsigned short*)(ws + 128 * MB);//  8 MB
  unsigned short* fin   = (unsigned short*)(ws + 136 * MB);//  8 MB
  unsigned short* hb    = (unsigned short*)(ws + 144 * MB);// 32 MB

  hipMemcpyAsync(lat, latents, (size_t)BATCH * NQ * DIM * sizeof(float),
                 hipMemcpyDeviceToDevice, stream);

  const float qscale = 0.08838834764831845f;  // DH^-0.5
  const int MLAT = BATCH * NQ;                // 2048
  const int MSEQ = BATCH * SEQ;               // 8192

  for (int l = 0; l < NLAYER; ++l) {
    // x = LN(seg + pos) -> bf16 ; lat_ln = LN(lat) -> bf16
    ln_kernel<true, false><<<MSEQ, 256, 0, stream>>>(
        seg, pos, nx_g + l * DIM, nx_b + l * DIM, xln);
    ln_kernel<false, false><<<MLAT, 256, 0, stream>>>(
        lat, nullptr, nl_g + l * DIM, nl_b + l * DIM, latln);
    // q = lat_ln @ Wq * scale  -> bf16 [2048, 2048]
    gemm_bf16w<EPI_BF16><<<dim3(DIM / 128, MLAT / 128), 256, 0, stream>>>(
        latln, DIM, Wq + (size_t)l * DIM * DIM, DIM, DIM, qscale,
        qb, nullptr, nullptr, nullptr, nullptr);
    // kv = x @ Wkv -> K [b,h,s,d] bf16 and V^T [b,h,d,s] bf16
    gemm_bf16w<EPI_KV><<<dim3(2 * DIM / 128, MSEQ / 128), 256, 0, stream>>>(
        xln, DIM, Wkv + (size_t)l * DIM * 2 * DIM, 2 * DIM, DIM, 1.f,
        nullptr, nullptr, nullptr, kbuf, vtbuf);
    // attention -> bf16 [2048, 2048]
    attn_flash<<<(BATCH * HEADS * (NQ / 16)) / 4, 128, 0, stream>>>(
        qb, kbuf, vtbuf, ao);
    // lat += attn_out @ Wo
    gemm_bf16w<EPI_F32ADD><<<dim3(DIM / 128, MLAT / 128), 256, 0, stream>>>(
        ao, DIM, Wo + (size_t)l * DIM * DIM, DIM, DIM, 1.f,
        nullptr, lat, lat, nullptr, nullptr);
    // FFN: h = silu(LN(lat) @ W1) ; lat += h @ W2
    ln_kernel<false, false><<<MLAT, 256, 0, stream>>>(
        lat, nullptr, fln_g + l * DIM, fln_b + l * DIM, fin);
    gemm_bf16w<EPI_SILU><<<dim3(FFD / 128, MLAT / 128), 256, 0, stream>>>(
        fin, DIM, W1 + (size_t)l * DIM * FFD, FFD, DIM, 1.f,
        hb, nullptr, nullptr, nullptr, nullptr);
    gemm_bf16w<EPI_F32ADD><<<dim3(DIM / 128, MLAT / 128), 256, 0, stream>>>(
        hb, FFD, W2 + (size_t)l * FFD * DIM, DIM, FFD, 1.f,
        nullptr, lat, lat, nullptr, nullptr);
  }
  // final LN -> fp32 output
  ln_kernel<false, true><<<MLAT, 256, 0, stream>>>(
      lat, nullptr, fn_g, fn_b, out);
}
